// NAFNetBlurCLIP_87763361727188
// MI455X (gfx1250) — matrix operand order repaired and checked by reading, not compiled, not run on hardware
//
#include <hip/hip_runtime.h>
#include <hip/hip_bf16.h>

// ---------------------------------------------------------------------------
// NAFNet-style block for MI455X (gfx1250, wave32).
// Memory-bound (~6.7 GFLOP vs ~100MB min traffic @23.3TB/s) -> bf16
// intermediates + v_wmma_f32_16x16x32_bf16 for all 1x1 convs, and TDM
// (tensor_load_to_lds + s_wait_tensorcnt) to stage the gated activations.
// ---------------------------------------------------------------------------

typedef __bf16 bf16;
typedef __attribute__((ext_vector_type(16))) __bf16 v16bf;
typedef __attribute__((ext_vector_type(8)))  __bf16 v8bf;
typedef __attribute__((ext_vector_type(8)))  float  v8f;
typedef __attribute__((ext_vector_type(4)))  unsigned int u32x4;
typedef __attribute__((ext_vector_type(8)))  unsigned int u32x8;

constexpr int HWPIX = 256 * 256;   // pixels per (b, channel) plane

// ---- WMMA fragment gathers from LDS (layouts per CDNA5 ISA 7.12.2) --------
// A (16x32 bf16): lane L -> row M = mtile*16 + (L&15), g = L>>4.
// VGPRs 0..7 hold K = 16g..16g+15 (contig, matches frag_b).
__device__ __forceinline__ v16bf frag_a(const bf16* w_lds, int mtile, int ktile, int lane) {
  int m = mtile * 16 + (lane & 15);
  int g = lane >> 4;
  const bf16* base = w_lds + m * 64 + ktile * 32 + g * 16;
  v8bf lo = *(const v8bf*)(base);        // K = kt*32 + 16g .. +7
  v8bf hi = *(const v8bf*)(base + 8);    // K = kt*32 + 16g + 8 .. +15
  v16bf a;
#pragma unroll
  for (int i = 0; i < 8; ++i) { a[i] = lo[i]; a[i + 8] = hi[i]; }
  return a;
}

// B (32x16 bf16): lane L -> col N = ntile*16 + (L&15), khalf = L>>4.
// VGPR v holds K = khalf*16 + 2v,2v+1 -> the 16 values are K-contiguous.
__device__ __forceinline__ v16bf frag_b(const bf16* x_lds, int ntile, int ktile, int lane) {
  int n  = ntile * 16 + (lane & 15);
  int kh = lane >> 4;
  return *(const v16bf*)(x_lds + n * 64 + ktile * 32 + kh * 16);
}

// ---- TDM: async 3-D tile load (cols x rows x channels) of bf16 into LDS ---
// D# per CDNA5 ISA: group0 {count/lds/global/type},
// group1 {data_size, tensor dims 0-1, tile dims 0-2, strides 0-1},
// group2 {tensor_dim2}, group3 {0}.
__device__ __forceinline__ void tdm_load_g_chunk(const bf16* gptr, unsigned lds_off,
                                                 int b, int u, int v, int chunk) {
  unsigned long long gaddr = (unsigned long long)(uintptr_t)gptr +
      2ull * ((unsigned long long)b * 64 * HWPIX +
              (unsigned long long)(u * 32 + chunk * 2) * 256 + (unsigned)(v * 32));
  u32x4 g0;
  g0[0] = 1u;                                   // count=1 (valid), user mode
  g0[1] = lds_off;                              // LDS byte address of tile
  g0[2] = (unsigned)gaddr;                      // global_addr[31:0]
  g0[3] = (unsigned)((gaddr >> 32) & 0x01FFFFFFu) | (2u << 30); // [56:32] | type=2
  u32x8 g1;
  g1[0] = (1u << 16);                           // wg_mask=0, data_size=1 (2 bytes)
  g1[1] = (256u & 0xFFFFu) << 16;               // tensor_dim0 lo16 (W=256)
  g1[2] = (256u >> 16) | ((256u & 0xFFFFu) << 16); // td0 hi16 | tensor_dim1 lo16 (H)
  g1[3] = (256u >> 16) | (32u << 16);           // td1 hi16 | tile_dim0 = 32 cols
  g1[4] = 2u | (64u << 16);                     // tile_dim1 = 2 rows | tile_dim2 = 64 ch
  g1[5] = 256u;                                 // tensor_dim0_stride lo32 (row stride)
  g1[6] = 0u;                                   // td0_stride hi16 | td1_stride lo16
  g1[7] = (unsigned)(65536u >> 16);             // td1_stride[47:16] (ch stride 65536)
  u32x4 g2; g2[0] = 128u; g2[1] = 0u; g2[2] = 0u; g2[3] = 0u;  // tensor_dim2 = B*C planes
  u32x4 g3; g3[0] = 0u;   g3[1] = 0u; g3[2] = 0u; g3[3] = 0u;
  asm volatile("tensor_load_to_lds %0, %1, %2, %3"
               :: "s"(g0), "s"(g1), "s"(g2), "s"(g3) : "memory");
}

// ===========================================================================
// Kernel A: layernorm1 + modulated conv1x1 (64 -> 128), WMMA. Block = cell.
// ===========================================================================
__global__ __launch_bounds__(256) void k_ln1_conv1(
    const float* __restrict__ x, const float* __restrict__ w1, const float* __restrict__ b1,
    const float* __restrict__ n1w, const float* __restrict__ n1b,
    const float* __restrict__ mod1, bf16* __restrict__ y)
{
  __shared__ __align__(32) bf16 w_lds[128 * 64];   // 16 KB, [o][c]
  __shared__ __align__(32) bf16 x_lds[256 * 64];   // 32 KB, [pix][c]
  int blk = blockIdx.x;
  int v = blk & 7, u = (blk >> 3) & 7, b = blk >> 6;
  int uv = u * 8 + v;
  int t = threadIdx.x, lane = t & 31, wave = t >> 5;

  // Modulated, row-normalized weights -> LDS bf16.
  if (t < 128) {
    int o = t; float tmp[64]; float ss = 0.f;
#pragma unroll
    for (int c = 0; c < 64; ++c) {
      float wv = w1[o * 64 + c] * mod1[((b * 128 + o) * 64 + c) * 64 + uv];
      tmp[c] = wv; ss += wv * wv;
    }
    float inv = rsqrtf(ss);
#pragma unroll
    for (int c = 0; c < 64; ++c) w_lds[o * 64 + c] = (bf16)(tmp[c] * inv);
  }

  for (int chunk = 0; chunk < 4; ++chunk) {         // 4 x 256 pixels = 32x32 patch
    __syncthreads();
    { // layernorm over channels for one pixel per thread
      int p  = t;
      int ph = chunk * 8 + (p >> 5), pw = p & 31;
      int gp = (u * 32 + ph) * 256 + (v * 32 + pw);
      const float* xb = x + (size_t)b * 64 * HWPIX + gp;
      if (chunk < 3) __builtin_prefetch(xb + 8 * 256, 0, 1);  // global_prefetch_b8
      float vals[64]; float mu = 0.f;
#pragma unroll
      for (int c = 0; c < 64; ++c) { float xv = xb[c * HWPIX]; vals[c] = xv; mu += xv; }
      mu *= (1.f / 64.f);
      float var = 0.f;
#pragma unroll
      for (int c = 0; c < 64; ++c) { float d = vals[c] - mu; var += d * d; }
      float inv = rsqrtf(var * (1.f / 64.f) + 1e-6f);
#pragma unroll
      for (int c = 0; c < 64; ++c)
        x_lds[p * 64 + c] = (bf16)((vals[c] - mu) * inv * n1w[c] + n1b[c]);
    }
    __syncthreads();
    // 8 mtiles x 16 ntiles = 128 tiles / 8 waves
    for (int tt = wave; tt < 128; tt += 8) {
      int mt = tt & 7, nt = tt >> 3;
      v8f acc = {};
#pragma unroll
      for (int kt = 0; kt < 2; ++kt)
        acc = __builtin_amdgcn_wmma_f32_16x16x32_bf16(
            false, frag_a(w_lds, mt, kt, lane),
            false, frag_b(x_lds, nt, kt, lane),
            (short)0, acc, false, false);
      int p  = nt * 16 + (lane & 15);
      int ph = chunk * 8 + (p >> 5), pw = p & 31;
      int gp = (u * 32 + ph) * 256 + (v * 32 + pw);
      int mb = mt * 16 + 8 * (lane >> 4);
#pragma unroll
      for (int r = 0; r < 8; ++r) {
        int oc = mb + r;
        y[(size_t)(b * 128 + oc) * HWPIX + gp] = (bf16)(acc[r] + b1[oc]);
      }
    }
  }
}

// ===========================================================================
// Kernel B: modulated depthwise 3x3 + SimpleGate + deterministic channel sums
// ===========================================================================
__global__ __launch_bounds__(256) void k_dw_gate(
    const bf16* __restrict__ y, const float* __restrict__ w2, const float* __restrict__ b2,
    const float* __restrict__ mod2, bf16* __restrict__ gbuf, float* __restrict__ partials)
{
  __shared__ float wk[128 * 9];
  __shared__ __align__(32) bf16 out_lds[256 * 64];
  __shared__ float acc4[4 * 64];
  int blk = blockIdx.x;
  int v = blk & 7, u = (blk >> 3) & 7, b = blk >> 6;
  int t = threadIdx.x;

  if (t < 128) {
    int ch = t;
    float m = mod2[((b * 128 + ch) * 8 + u) * 8 + v];
    float tmp[9]; float ss = 0.f;
#pragma unroll
    for (int i = 0; i < 9; ++i) { float wv = w2[ch * 9 + i] * m; tmp[i] = wv; ss += wv * wv; }
    float inv = rsqrtf(ss);
#pragma unroll
    for (int i = 0; i < 9; ++i) wk[ch * 9 + i] = tmp[i] * inv;
  }
  { int q = t >> 6, c = t & 63; acc4[q * 64 + c] = 0.f; }
  __syncthreads();

  for (int chunk = 0; chunk < 4; ++chunk) {
    int p  = t;
    int ph = chunk * 8 + (p >> 5), pw = p & 31;
    int h = u * 32 + ph, w_ = v * 32 + pw;
    for (int cg = 0; cg < 64; ++cg) {
      float a = b2[cg], q = b2[cg + 64];
      const bf16* ya = y + (size_t)(b * 128 + cg) * HWPIX;
      const bf16* yq = y + (size_t)(b * 128 + cg + 64) * HWPIX;
      const float* wa = &wk[cg * 9];
      const float* wq = &wk[(cg + 64) * 9];
#pragma unroll
      for (int dh = -1; dh <= 1; ++dh) {
        int hh = h + dh; if ((unsigned)hh >= 256u) continue;
#pragma unroll
        for (int dw = -1; dw <= 1; ++dw) {
          int ww = w_ + dw; if ((unsigned)ww >= 256u) continue;
          int ki = (dh + 1) * 3 + (dw + 1);
          a += (float)ya[hh * 256 + ww] * wa[ki];
          q += (float)yq[hh * 256 + ww] * wq[ki];
        }
      }
      float o = a * q;
      out_lds[p * 64 + cg] = (bf16)o;
      gbuf[(size_t)(b * 64 + cg) * HWPIX + h * 256 + w_] = (bf16)o;
    }
    __syncthreads();
    { // deterministic reduction: thread owns (quarter, channel) slot
      int cg = t & 63, qq = t >> 6;
      float sacc = 0.f;
      for (int pp = 0; pp < 64; ++pp) sacc += (float)out_lds[(qq * 64 + pp) * 64 + cg];
      acc4[qq * 64 + cg] += sacc;
    }
    __syncthreads();
  }
  if (t < 64)
    partials[blk * 64 + t] = acc4[t] + acc4[64 + t] + acc4[128 + t] + acc4[192 + t];
}

// ===========================================================================
// Kernel C: SCA  s[b,o] = sca_b[o] + sum_c sca_w[o,c] * mean[b,c]
// ===========================================================================
__global__ void k_sca(const float* __restrict__ partials, const float* __restrict__ sca_w,
                      const float* __restrict__ sca_b, float* __restrict__ s)
{
  __shared__ float smean[128];
  int t = threadIdx.x;          // 128 threads
  {
    int b = t >> 6, c = t & 63;
    float m = 0.f;
    for (int cell = 0; cell < 64; ++cell) m += partials[(b * 64 + cell) * 64 + c];
    smean[t] = m * (1.f / (float)HWPIX);
  }
  __syncthreads();
  int b = t >> 6, o = t & 63;
  float accv = 0.f;
  for (int c = 0; c < 64; ++c) accv += sca_w[o * 64 + c] * smean[b * 64 + c];
  s[t] = accv + sca_b[o];
}

// ===========================================================================
// Kernel D: fused tail — TDM-staged g, s-scale, conv3(WMMA)+beta-residual,
//           ln2, conv4(WMMA), gate, conv5(WMMA)+gamma-residual. 64 KB LDS.
// TDM for chunk k+1 is issued as soon as chunk k's staging buffer is
// consumed, overlapping the DMA with the three WMMA stages.
// ===========================================================================
__global__ __launch_bounds__(256) void k_fused_tail(
    const float* __restrict__ x, const bf16* __restrict__ g, const float* __restrict__ s,
    const float* __restrict__ w3, const float* __restrict__ b3, const float* __restrict__ mod3,
    const float* __restrict__ w4, const float* __restrict__ b4, const float* __restrict__ mod4,
    const float* __restrict__ w5, const float* __restrict__ b5, const float* __restrict__ mod5,
    const float* __restrict__ n2w, const float* __restrict__ n2b,
    const float* __restrict__ beta, const float* __restrict__ gam,
    float* __restrict__ out)
{
  __shared__ __align__(32) bf16 w3_lds[64 * 64];     // 8 KB
  __shared__ __align__(32) bf16 w4_lds[128 * 64];    // 16 KB
  __shared__ __align__(32) bf16 bufA[64 * 64];       // 8 KB: g*s -> ln2out -> gated
  __shared__ __align__(32) bf16 bufB[64 * 64];       // 8 KB: y1
  __shared__ __align__(32) char smemC[64 * 128 * 2]; // 16 KB: conv4 out | w5 (aliased)
  __shared__ __align__(32) bf16 rawg[64 * 64];       // 8 KB: TDM tile [ch][2*32 pix]
  bf16* bufC   = (bf16*)smemC;
  bf16* w5_lds = (bf16*)smemC;  // valid only after gate consumed bufC

  int blk = blockIdx.x;
  int v = blk & 7, u = (blk >> 3) & 7, b = blk >> 6;
  int uv = u * 8 + v;
  int t = threadIdx.x, lane = t & 31, wave = t >> 5;
  // LDS byte offset of rawg: flat __shared__ address keeps offset in [31:0].
  unsigned raw_off = (unsigned)(uintptr_t)(&rawg[0]);

  if (wave == 0) tdm_load_g_chunk(g, raw_off, b, u, v, 0);   // prime pipeline

  if (t < 64) {   // w3 (64x64)
    int o = t; float tmp[64]; float ss = 0.f;
#pragma unroll
    for (int c = 0; c < 64; ++c) {
      float wv = w3[o * 64 + c] * mod3[((b * 64 + o) * 64 + c) * 64 + uv];
      tmp[c] = wv; ss += wv * wv;
    }
    float inv = rsqrtf(ss);
#pragma unroll
    for (int c = 0; c < 64; ++c) w3_lds[o * 64 + c] = (bf16)(tmp[c] * inv);
  }
  if (t < 128) {  // w4 (128x64)
    int o = t; float tmp[64]; float ss = 0.f;
#pragma unroll
    for (int c = 0; c < 64; ++c) {
      float wv = w4[o * 64 + c] * mod4[((b * 128 + o) * 64 + c) * 64 + uv];
      tmp[c] = wv; ss += wv * wv;
    }
    float inv = rsqrtf(ss);
#pragma unroll
    for (int c = 0; c < 64; ++c) w4_lds[o * 64 + c] = (bf16)(tmp[c] * inv);
  }

  for (int chunk = 0; chunk < 16; ++chunk) {      // 16 x 64 pixels (2 rows)
    __builtin_amdgcn_s_wait_tensorcnt(0);         // rawg ready (no-op for waves 1..7)
    __syncthreads();
    { // stage 1: bufA[pix][c] = rawg[c][pix] * s[c]   (LDS transpose + scale)
      int p = t & 63, cb = t >> 6;
#pragma unroll
      for (int cc = cb; cc < 64; cc += 4)
        bufA[p * 64 + cc] = (bf16)((float)rawg[cc * 64 + p] * s[b * 64 + cc]);
    }
    __syncthreads();
    if (wave == 0 && chunk + 1 < 16)              // overlap next DMA with WMMA work
      tdm_load_g_chunk(g, raw_off, b, u, v, chunk + 1);
    // conv3 + beta residual -> bufB (y1).  4x4 tiles / 8 waves.
    for (int tt = wave; tt < 16; tt += 8) {
      int mt = tt & 3, nt = tt >> 2;
      v8f acc = {};
#pragma unroll
      for (int kt = 0; kt < 2; ++kt)
        acc = __builtin_amdgcn_wmma_f32_16x16x32_bf16(
            false, frag_a(w3_lds, mt, kt, lane),
            false, frag_b(bufA, nt, kt, lane),
            (short)0, acc, false, false);
      int p  = nt * 16 + (lane & 15);
      int ph = chunk * 2 + (p >> 5), pw = p & 31;
      int gp = (u * 32 + ph) * 256 + (v * 32 + pw);
      int mb = mt * 16 + 8 * (lane >> 4);
#pragma unroll
      for (int r = 0; r < 8; ++r) {
        int oc = mb + r;
        float y1 = x[(size_t)(b * 64 + oc) * HWPIX + gp] + (acc[r] + b3[oc]) * beta[oc];
        bufB[p * 64 + oc] = (bf16)y1;
      }
    }
    __syncthreads();
    // ln2 -> bufA
    if (t < 64) {
      int p = t; float vals[64]; float mu = 0.f;
#pragma unroll
      for (int c = 0; c < 64; ++c) { float xv = (float)bufB[p * 64 + c]; vals[c] = xv; mu += xv; }
      mu *= (1.f / 64.f);
      float var = 0.f;
#pragma unroll
      for (int c = 0; c < 64; ++c) { float d = vals[c] - mu; var += d * d; }
      float inv = rsqrtf(var * (1.f / 64.f) + 1e-6f);
#pragma unroll
      for (int c = 0; c < 64; ++c)
        bufA[p * 64 + c] = (bf16)((vals[c] - mu) * inv * n2w[c] + n2b[c]);
    }
    __syncthreads();
    // conv4 -> bufC (128 ch).  8x4 tiles / 8 waves.
    for (int tt = wave; tt < 32; tt += 8) {
      int mt = tt & 7, nt = tt >> 3;
      v8f acc = {};
#pragma unroll
      for (int kt = 0; kt < 2; ++kt)
        acc = __builtin_amdgcn_wmma_f32_16x16x32_bf16(
            false, frag_a(w4_lds, mt, kt, lane),
            false, frag_b(bufA, nt, kt, lane),
            (short)0, acc, false, false);
      int p  = nt * 16 + (lane & 15);
      int mb = mt * 16 + 8 * (lane >> 4);
#pragma unroll
      for (int r = 0; r < 8; ++r) {
        int oc = mb + r;
        bufC[p * 128 + oc] = (bf16)(acc[r] + b4[oc]);
      }
    }
    __syncthreads();
    { // gate -> bufA (ln2 content consumed by conv4 already)
      int p = t & 63, cb = t >> 6;
#pragma unroll
      for (int cc = cb; cc < 64; cc += 4)
        bufA[p * 64 + cc] = (bf16)((float)bufC[p * 128 + cc] * (float)bufC[p * 128 + cc + 64]);
    }
    __syncthreads();
    // w5 (aliases bufC) — recompute after gate consumed bufC
    if (t < 64) {
      int o = t; float tmp[64]; float ss = 0.f;
#pragma unroll
      for (int c = 0; c < 64; ++c) {
        float wv = w5[o * 64 + c] * mod5[((b * 64 + o) * 64 + c) * 64 + uv];
        tmp[c] = wv; ss += wv * wv;
      }
      float inv = rsqrtf(ss);
#pragma unroll
      for (int c = 0; c < 64; ++c) w5_lds[o * 64 + c] = (bf16)(tmp[c] * inv);
    }
    __syncthreads();
    // conv5 + gamma residual -> out (fp32)
    for (int tt = wave; tt < 16; tt += 8) {
      int mt = tt & 3, nt = tt >> 2;
      v8f acc = {};
#pragma unroll
      for (int kt = 0; kt < 2; ++kt)
        acc = __builtin_amdgcn_wmma_f32_16x16x32_bf16(
            false, frag_a(w5_lds, mt, kt, lane),
            false, frag_b(bufA, nt, kt, lane),
            (short)0, acc, false, false);
      int p  = nt * 16 + (lane & 15);
      int ph = chunk * 2 + (p >> 5), pw = p & 31;
      int gp = (u * 32 + ph) * 256 + (v * 32 + pw);
      int mb = mt * 16 + 8 * (lane >> 4);
#pragma unroll
      for (int r = 0; r < 8; ++r) {
        int oc = mb + r;
        out[(size_t)(b * 64 + oc) * HWPIX + gp] =
            (float)bufB[p * 64 + oc] + (acc[r] + b5[oc]) * gam[oc];
      }
    }
  }
}

// ===========================================================================
extern "C" void kernel_launch(void* const* d_in, const int* in_sizes, int n_in,
                              void* d_out, int out_size, void* d_ws, size_t ws_size,
                              hipStream_t stream) {
  (void)in_sizes; (void)n_in; (void)out_size; (void)ws_size;
  const float* x    = (const float*)d_in[0];
  const float* w1   = (const float*)d_in[1];
  const float* b1   = (const float*)d_in[2];
  const float* w2   = (const float*)d_in[3];
  const float* b2   = (const float*)d_in[4];
  const float* w3   = (const float*)d_in[5];
  const float* b3   = (const float*)d_in[6];
  const float* scaw = (const float*)d_in[7];
  const float* scab = (const float*)d_in[8];
  const float* w4   = (const float*)d_in[9];
  const float* b4   = (const float*)d_in[10];
  const float* w5   = (const float*)d_in[11];
  const float* b5   = (const float*)d_in[12];
  const float* n1w  = (const float*)d_in[13];
  const float* n1b  = (const float*)d_in[14];
  const float* n2w  = (const float*)d_in[15];
  const float* n2b  = (const float*)d_in[16];
  const float* beta = (const float*)d_in[17];
  const float* gam  = (const float*)d_in[18];
  const float* mod1 = (const float*)d_in[19];
  const float* mod2 = (const float*)d_in[20];
  const float* mod3 = (const float*)d_in[21];
  const float* mod4 = (const float*)d_in[22];
  const float* mod5 = (const float*)d_in[23];

  char* ws = (char*)d_ws;
  bf16* y = (bf16*)ws;                 size_t off = (size_t)2 * 128 * HWPIX * 2; // 32 MB
  bf16* g = (bf16*)(ws + off);         off += (size_t)2 * 64 * HWPIX * 2;        // 16 MB
  float* partials = (float*)(ws + off); off += (size_t)128 * 64 * 4;             // 32 KB
  float* s = (float*)(ws + off);

  k_ln1_conv1 <<<128, 256, 0, stream>>>(x, w1, b1, n1w, n1b, mod1, y);
  k_dw_gate   <<<128, 256, 0, stream>>>(y, w2, b2, mod2, g, partials);
  k_sca       <<<1,   128, 0, stream>>>(partials, scaw, scab, s);
  k_fused_tail<<<128, 256, 0, stream>>>(x, g, s, w3, b3, mod3, w4, b4, mod4,
                                        w5, b5, mod5, n2w, n2b, beta, gam,
                                        (float*)d_out);
}
